// Representation_58961311040331
// MI455X (gfx1250) — matrix operand order repaired and checked by reading, not compiled, not run on hardware
//
#include <hip/hip_runtime.h>
#include <hip/hip_bf16.h>
#include <stdint.h>

// ---------------------------------------------------------------------------
// ConvLSTM2D forward, MI455X (gfx1250), wave32 + v_wmma_f32_16x16x32_f16.
// Shapes: B=8, T=16, H=64, W=64, Cin=32, F=64 (4F=256 gate channels).
// Implicit GEMM per step: M=32768 (B*H*W), N=256 (4F), K=864 (9*32 + 9*64).
// ---------------------------------------------------------------------------

typedef __attribute__((ext_vector_type(16))) _Float16 v16h;
typedef __attribute__((ext_vector_type(8)))  _Float16 v8h;
typedef __attribute__((ext_vector_type(8)))  float    v8f;
typedef __attribute__((ext_vector_type(4)))  float    v4f;

#define NB    8         // batch
#define NT    16        // timesteps
#define NH    64        // height
#define NW    64        // width
#define CIN   32
#define NF    64        // hidden channels
#define NGATE 256       // 4*F
#define KTOT  864       // 9*CIN + 9*NF
#define KBLK  27        // 864 / 32
#define KXB   9         // 288 / 32   (x part)
#define MROWS (NB*NH*NW)            // 32768
#define HN    (MROWS*NF)            // 2097152 elements of h / c

// ---------------------------------------------------------------------------
// Pre-pack weights (f32 W[3][3][32][256] ++ U[3][3][64][256]) into the f16
// WMMA B-fragment layout:
//   Bp[((kb*16 + ntile)*32 + lane)*16 + i] = src[(kb*32 + k(lane,i))*256 + ntile*16 + lane%16]
// Per ISA B layout: lane%16 = N col, k = 8*(lane/16) + i (i<8), 16 + 8*(lane/16) + i-8 (i>=8).
// ---------------------------------------------------------------------------
__global__ void __launch_bounds__(256)
prepack_weights(const float* __restrict__ Wf, const float* __restrict__ Uf,
                _Float16* __restrict__ Bp) {
    int idx = blockIdx.x * 256 + threadIdx.x;      // 864*256 = 221184 total
    if (idx >= KTOT * NGATE) return;
    int i     = idx & 15;
    int laneK = (idx >> 4) & 31;
    int ntile = (idx >> 9) & 15;
    int kb    = idx >> 13;                         // 0..26
    int K = kb * 32 + ((i & 8) << 1) + ((laneK >> 4) << 3) + (i & 7); // 0..863
    int n = ntile * 16 + (laneK & 15);             // 0..255
    float v = (K < 288) ? Wf[K * NGATE + n] : Uf[(K - 288) * NGATE + n];
    Bp[idx] = (_Float16)v;
}

// Zero initial hidden state (f16) and cell state (f32).
__global__ void __launch_bounds__(256)
init_state(_Float16* __restrict__ h0, float* __restrict__ c0) {
    int idx = blockIdx.x * 256 + threadIdx.x;
    if (idx < HN) { h0[idx] = (_Float16)0.0f; c0[idx] = 0.0f; }
}

__device__ __forceinline__ float hard_sigmoid(float x) {
    // jax.nn.hard_sigmoid = relu6(x+3)/6
    return fminf(fmaxf(x + 3.0f, 0.0f), 6.0f) * (1.0f / 6.0f);
}

// Gather one 16x32 f16 A fragment (ISA layout: lane = M%16, K halves split at
// lane 16). kb is a compile-time constant after full unroll, so all (ky,kx,
// chalf) decode folds away; only the per-lane bounds compares remain.
__device__ __forceinline__ v16h
gather_A(int kb, int t, int b, int y, int xw, int half,
         const float* __restrict__ x, const _Float16* __restrict__ h_prev) {
    v16h a = {};
    if (kb < KXB) {
        // x-patch part: kb -> (ky,kx), channels 0..31 contiguous in NHWC
        const int ky = kb / 3 - 1, kx = kb % 3 - 1;
        const int yy = y + ky, xx = xw + kx;
        if (yy >= 0 && yy < NH && xx >= 0 && xx < NW) {
            const float* px = x + ((((size_t)b * NT + t) * NH + yy) * NW + xx) * CIN;
            const int c0 = half * 8;               // K chunk A: [c0, c0+8)
            const v4f* pA = (const v4f*)(px + c0);
            const v4f* pB = (const v4f*)(px + 16 + c0);
            v4f f0 = pA[0], f1 = pA[1], f2 = pB[0], f3 = pB[1];
#pragma unroll
            for (int i = 0; i < 4; ++i) {
                a[i]      = (_Float16)f0[i];
                a[4 + i]  = (_Float16)f1[i];
                a[8 + i]  = (_Float16)f2[i];
                a[12 + i] = (_Float16)f3[i];
            }
        }
    } else {
        // h-patch part: 2 K-blocks per (ky,kx) over 64 hidden channels
        const int kbh   = kb - KXB;                // 0..17
        const int ky    = kbh / 6 - 1;
        const int kx    = (kbh % 6) / 2 - 1;
        const int chalf = kbh & 1;                 // which 32-channel half
        const int yy = y + ky, xx = xw + kx;
        if (yy >= 0 && yy < NH && xx >= 0 && xx < NW) {
            const _Float16* ph = h_prev +
                (((size_t)b * NH + yy) * NW + xx) * NF + chalf * 32;
            v8h cA = *(const v8h*)(ph + half * 8);
            v8h cB = *(const v8h*)(ph + 16 + half * 8);
#pragma unroll
            for (int i = 0; i < 8; ++i) { a[i] = cA[i]; a[8 + i] = cB[i]; }
        }
    }
    return a;
}

// ---------------------------------------------------------------------------
// One ConvLSTM step. Block = 128 threads = 4 waves.
//   wave(mwave=w>>1, nhalf=w&1): rows [blk*32 + mwave*16, +16),
//                                cols [nhalf*128, +128) of z = im2col(x_t,h)·B.
// Fully-unrolled K loop (27 blocks of 32), software-pipelined A gather,
// 8 B fragments held live per iteration so the 16 b128 loads clause up and
// overlap the 8-WMMA burst. Gates fused through LDS.
// ---------------------------------------------------------------------------
__global__ void __launch_bounds__(128)
convlstm_step(const float* __restrict__ x, const _Float16* __restrict__ Bp,
              const _Float16* __restrict__ h_prev, _Float16* __restrict__ h_next,
              float* __restrict__ c_state, const float* __restrict__ bias,
              float* __restrict__ out, int t, int write_out) {
    __shared__ float Zs[32][NGATE + 4];            // +4 pad: stagger banks

    const int tid   = threadIdx.x;
    const int wave  = tid >> 5;
    const int lane  = tid & 31;
    const int mwave = wave >> 1;                   // 0..1
    const int nhalf = wave & 1;                    // 0..1
    const int mlane = lane & 15;                   // row within 16-tile
    const int half  = lane >> 4;                   // K-half selector (A layout)

    const int rBase = blockIdx.x * 32;             // 32 spatial rows per block
    const int r  = rBase + mwave * 16 + mlane;     // this lane's output row
    const int b  = r >> 12;                        // r / (64*64)
    const int y  = (r >> 6) & 63;
    const int xw = r & 63;

    v8f acc[8] = {{}, {}, {}, {}, {}, {}, {}, {}};

    // Per-lane base into the packed B fragments for this wave's N-half.
    const v16h* __restrict__ Bv = (const v16h*)Bp + (size_t)nhalf * 8 * 32 + lane;

    // Software pipeline: A fragment for kb=0 up front.
    v16h a = gather_A(0, t, b, y, xw, half, x, h_prev);

#pragma unroll
    for (int kb = 0; kb < KBLK; ++kb) {
        // ---- issue all 16 b128 loads for this K-block's 8 B fragments
        v16h bf[8];
        const v16h* __restrict__ Bkb = Bv + (size_t)kb * 16 * 32;
#pragma unroll
        for (int nt = 0; nt < 8; ++nt) bf[nt] = Bkb[(size_t)nt * 32];

        // ---- gather next A fragment while WMMAs of this block execute
        v16h an = (kb + 1 < KBLK)
                      ? gather_A(kb + 1, t, b, y, xw, half, x, h_prev)
                      : a;

        // ---- 8 WMMA per K-block: this wave's 128 N columns
#pragma unroll
        for (int nt = 0; nt < 8; ++nt) {
            acc[nt] = __builtin_amdgcn_wmma_f32_16x16x32_f16(
                false, a, false, bf[nt], (short)0, acc[nt], false, false);
        }
        a = an;
    }

    // ---- stage z tile to LDS (C/D layout: M over VGPRs (+8 for hi lanes),
    //      N = lane%16 within each 16-wide subtile)
#pragma unroll
    for (int nt = 0; nt < 8; ++nt) {
#pragma unroll
        for (int rr = 0; rr < 8; ++rr) {
            Zs[mwave * 16 + rr + half * 8][nhalf * 128 + nt * 16 + mlane] =
                acc[nt][rr];
        }
    }
    __syncthreads();

    // ---- fused LSTM gate math: 32 rows x 64 F = 2048 cells, 16 per thread.
    //      f = tid & 63 is iteration-invariant -> hoist the 4 bias loads.
    const int f   = tid & 63;
    const int m0  = tid >> 6;
    const float b_i = bias[f];
    const float b_f = bias[64 + f];
    const float b_c = bias[128 + f];
    const float b_o = bias[192 + f];
#pragma unroll
    for (int it = 0; it < 16; ++it) {
        const int m = m0 + it * 2;                 // 0..31
        const float zi = Zs[m][f]       + b_i;
        const float zf = Zs[m][64 + f]  + b_f;
        const float zc = Zs[m][128 + f] + b_c;
        const float zo = Zs[m][192 + f] + b_o;
        const float ig = hard_sigmoid(zi);
        const float fg = hard_sigmoid(zf);
        const float og = hard_sigmoid(zo);
        const size_t g = (size_t)(rBase + m) * NF + f;
        const float c_old = c_state[g];
        const float c_new = fg * c_old + ig * fmaxf(zc, 0.0f);
        const float h_new = og * fmaxf(c_new, 0.0f);
        c_state[g] = c_new;
        h_next[g]  = (_Float16)h_new;
        if (write_out) out[g] = h_new;             // only on final timestep
    }
}

// ---------------------------------------------------------------------------
// Host launcher. Workspace layout (bytes):
//   [0x000000, 0x06C000)  packed f16 weights (864*256*2 = 442368)
//   [0x06C000, 0x46C000)  h buffer A (f16, 4 MB)
//   [0x46C000, 0x86C000)  h buffer B (f16, 4 MB)
//   [0x86C000, 0x106C000) c state   (f32, 8 MB)
// ---------------------------------------------------------------------------
extern "C" void kernel_launch(void* const* d_in, const int* in_sizes, int n_in,
                              void* d_out, int out_size, void* d_ws, size_t ws_size,
                              hipStream_t stream) {
    (void)in_sizes; (void)n_in; (void)out_size; (void)ws_size;

    const float* x    = (const float*)d_in[0];   // (8,16,64,64,32)
    const float* Wf   = (const float*)d_in[1];   // (3,3,32,256)
    const float* Uf   = (const float*)d_in[2];   // (3,3,64,256)
    const float* bias = (const float*)d_in[3];   // (256,)
    float* out = (float*)d_out;                  // (8,64,64,64)

    char* ws = (char*)d_ws;
    _Float16* Bp = (_Float16*)(ws);
    _Float16* hA = (_Float16*)(ws + 0x06C000);
    _Float16* hB = (_Float16*)(ws + 0x46C000);
    float*    cS = (float*)   (ws + 0x86C000);

    prepack_weights<<<(KTOT * NGATE + 255) / 256, 256, 0, stream>>>(Wf, Uf, Bp);
    init_state<<<(HN + 255) / 256, 256, 0, stream>>>(hA, cS);

    for (int t = 0; t < NT; ++t) {
        const _Float16* hp = (t & 1) ? hB : hA;
        _Float16*       hn = (t & 1) ? hA : hB;
        convlstm_step<<<MROWS / 32, 128, 0, stream>>>(
            x, Bp, hp, hn, cS, bias, out, t, (t == NT - 1) ? 1 : 0);
    }
}
